// AttentionLayer_42537356099848
// MI455X (gfx1250) — matrix operand order repaired and checked by reading, not compiled, not run on hardware
//
#include <hip/hip_runtime.h>
#include <hip/hip_bf16.h>
#include <stdint.h>

// Problem constants (match reference)
#define B_   2
#define SQ_  2048
#define SC_  2048
#define SKV_ 4096
#define DIM_ 1024
#define H_   16
#define HD_  64
#define NEG_ (-10000.0f)

typedef __attribute__((ext_vector_type(16))) __bf16 bf16x16;
typedef __attribute__((ext_vector_type(8)))  float  f32x8;

// Fragment union: 16 bf16 = 32 bytes = two uint4 (ds_load_b128 reads).
union Frag { bf16x16 v; unsigned short u[16]; uint4 q[2]; };

__device__ __forceinline__ unsigned short f32_to_bf16(float f) {
    union { float f; uint32_t u; } c; c.f = f;
    uint32_t u = c.u;
    u += 0x7FFFu + ((u >> 16) & 1u);      // round-to-nearest-even
    return (unsigned short)(u >> 16);
}

// Packed f32x2 -> bf16x2 store (4B-aligned dst). Uses v_cvt_pk_bf16_f32 when
// the builtin exists on this toolchain, else manual RNE.
#if __has_builtin(__builtin_amdgcn_cvt_pk_bf16_f32)
typedef __attribute__((ext_vector_type(2))) __bf16 bf16x2;
__device__ __forceinline__ void cvt2_store(unsigned short* d, float x, float y) {
    union { bf16x2 v; unsigned u; } c;
    c.v = __builtin_amdgcn_cvt_pk_bf16_f32(x, y);
    *(unsigned*)d = c.u;
}
#else
__device__ __forceinline__ void cvt2_store(unsigned short* d, float x, float y) {
    d[0] = f32_to_bf16(x); d[1] = f32_to_bf16(y);
}
#endif

// Low 32 bits of a generic pointer to LDS == LDS byte offset (aperture rule).
__device__ __forceinline__ unsigned lds_off(const void* p) {
    return (unsigned)(size_t)p;
}

// Async global->LDS 16-byte copy (CDNA5, tracked by ASYNCcnt).
__device__ __forceinline__ void async_copy_b128(unsigned ldsoff, const void* g) {
    asm volatile("global_load_async_to_lds_b128 %0, %1, off"
                 :: "v"(ldsoff), "v"((unsigned long long)g) : "memory");
}
__device__ __forceinline__ void wait_async0() {
    asm volatile("s_wait_asynccnt 0x0" ::: "memory");
}

// ---------------------------------------------------------------------------
// GEMM: C[M,N] = A[M,K] @ W[K,N] + bias[N]. fp32 in memory, bf16 WMMA compute.
// 128 thr = 4 waves, block tile 128x64, wave tile 32x64, K-step 32,
// LDS double-buffered (stage k+1 while computing k; one barrier per step).
// A-rows per batch are SQ_ (2048); concat remap uses shift/mask.
// ---------------------------------------------------------------------------
#define GT_M 128
#define GT_N 64
#define GT_K 32
#define AP   40   // padded row stride (ushorts), multiple of 8 -> 16B aligned

__device__ __forceinline__ void gemm_stage(
    const float* __restrict__ A, const float* __restrict__ W,
    int blockM, int blockN, int K, int N, int k0, int tid,
    unsigned short (*__restrict__ as)[AP], unsigned short (*__restrict__ bsT)[AP])
{
    float4 fa[8], fw[4];
    #pragma unroll
    for (int g = 0; g < 8; ++g) {       // A tile 128x32: 1024 float4 groups
        int idx4 = tid + g * 128;
        int r = idx4 >> 3, c4 = (idx4 & 7) * 4;
        fa[g] = *(const float4*)&A[(size_t)(blockM + r) * K + k0 + c4];
    }
    #pragma unroll
    for (int g = 0; g < 4; ++g) {       // W tile 32x64: 512 float4 groups
        int idx4 = tid + g * 128;
        int r = idx4 >> 4, c4 = (idx4 & 15) * 4;
        fw[g] = *(const float4*)&W[(size_t)(k0 + r) * N + blockN + c4];
    }
    #pragma unroll
    for (int g = 0; g < 8; ++g) {
        int idx4 = tid + g * 128;
        int r = idx4 >> 3, c4 = (idx4 & 7) * 4;
        unsigned short* d = &as[r][c4];
        cvt2_store(d,     fa[g].x, fa[g].y);
        cvt2_store(d + 2, fa[g].z, fa[g].w);
    }
    #pragma unroll
    for (int g = 0; g < 4; ++g) {       // transposed scatter into BsT[col][k]
        int idx4 = tid + g * 128;
        int r = idx4 >> 4, c4 = (idx4 & 15) * 4;
        bsT[c4 + 0][r] = f32_to_bf16(fw[g].x);
        bsT[c4 + 1][r] = f32_to_bf16(fw[g].y);
        bsT[c4 + 2][r] = f32_to_bf16(fw[g].z);
        bsT[c4 + 3][r] = f32_to_bf16(fw[g].w);
    }
}

__global__ __launch_bounds__(128)
void gemm_bf16_wmma(const float* __restrict__ A, const float* __restrict__ W,
                    const float* __restrict__ bias,
                    float* __restrict__ C, unsigned short* __restrict__ Cbf,
                    int M, int N, int K,
                    int mOutPerB, int rowBase, int vTrans)
{
    __shared__ __align__(16) unsigned short As[2][GT_M][AP];
    __shared__ __align__(16) unsigned short BsT[2][GT_N][AP];

    const int tid  = threadIdx.x;
    const int lane = tid & 31;
    const int wave = tid >> 5;          // 0..3 -> 32 rows each
    const int blockM = blockIdx.y * GT_M;
    const int blockN = blockIdx.x * GT_N;
    const int hlf = lane >> 4;
    const int l15 = lane & 15;

    f32x8 acc[2][4] = {};               // wave tile 32x64 = 2x4 WMMA tiles

    gemm_stage(A, W, blockM, blockN, K, N, 0, tid, As[0], BsT[0]);
    __syncthreads();

    int p = 0;
    for (int k0 = 0; k0 < K; k0 += GT_K) {
        if (k0 + GT_K < K) {
            gemm_stage(A, W, blockM, blockN, K, N, k0 + GT_K, tid,
                       As[p ^ 1], BsT[p ^ 1]);
            __builtin_prefetch(&A[(size_t)(blockM + (tid >> 3)) * K + k0 + 2 * GT_K], 0, 1);
        }

        Frag af[2], bf[4];
        #pragma unroll
        for (int mr = 0; mr < 2; ++mr) {
            const unsigned short* arow = &As[p][wave * 32 + mr * 16 + l15][0];
            af[mr].q[0] = *(const uint4*)(arow + 8 * hlf);
            af[mr].q[1] = *(const uint4*)(arow + 16 + 8 * hlf);
        }
        #pragma unroll
        for (int nr = 0; nr < 4; ++nr) {
            const unsigned short* bcol = &BsT[p][nr * 16 + l15][0];
            bf[nr].q[0] = *(const uint4*)(bcol + 8 * hlf);
            bf[nr].q[1] = *(const uint4*)(bcol + 16 + 8 * hlf);
        }
        #pragma unroll
        for (int mr = 0; mr < 2; ++mr)
            #pragma unroll
            for (int nr = 0; nr < 4; ++nr)
                acc[mr][nr] = __builtin_amdgcn_wmma_f32_16x16x32_bf16(
                    false, af[mr].v, false, bf[nr].v,
                    (short)0, acc[mr][nr], false, false);
        __syncthreads();
        p ^= 1;
    }

    // Epilogue: C layout VGPR i -> row = i + 8*(lane>>4), col = lane&15.
    // Per-batch input span is SQ_ = 2048 (power of two) -> shift/mask remap.
    #pragma unroll
    for (int mr = 0; mr < 2; ++mr) {
        #pragma unroll
        for (int nr = 0; nr < 4; ++nr) {
            #pragma unroll
            for (int i = 0; i < 8; ++i) {
                int row = blockM + wave * 32 + mr * 16 + i + 8 * hlf;
                int col = blockN + nr * 16 + l15;
                int bb = row >> 11;            // / SQ_
                int rr = row & (SQ_ - 1);      // % SQ_
                float val = acc[mr][nr][i] + bias[col];
                size_t orow = (size_t)bb * mOutPerB + rowBase + rr;
                if (C) C[orow * (size_t)N + col] = val;
                if (Cbf) {
                    if (!vTrans) {
                        Cbf[orow * (size_t)N + col] = f32_to_bf16(val);
                    } else {  // [B,H,HD,SKV] head-transposed layout for V
                        int hh = col >> 6, dd = col & 63;
                        Cbf[(((size_t)bb * H_ + hh) * HD_ + dd) * SKV_ +
                            (rowBase + rr)] = f32_to_bf16(val);
                    }
                }
            }
        }
    }
}

// ---------------------------------------------------------------------------
// Flash attention, all-bf16 operands, double-buffered async-LDS K/V staging.
//   qb : [B,SQ,DIM] bf16    kb : [B,SKV,DIM] bf16   vbT: [B,H,HD,SKV] bf16
// Block = 4 waves, each wave owns 16 q rows of one (b,h).
// ---------------------------------------------------------------------------
#define AT_QT 64
#define AT_KT 32
#define KSP   72   // Ks row stride (64 + 8 pad), 16B multiple

__device__ __forceinline__ void attn_stage(
    const unsigned short* __restrict__ kb, const unsigned short* __restrict__ vbT,
    int b, int h, int kv0, int tid, unsigned ksb, unsigned vsb)
{
    #pragma unroll
    for (int g = 0; g < 2; ++g) {       // K tile: 32 rows x 128B = 256 chunks
        int ch = tid + g * 128;
        int r = ch >> 3, seg = ch & 7;
        const unsigned short* gp =
            kb + ((size_t)b * SKV_ + kv0 + r) * DIM_ + h * HD_ + seg * 8;
        async_copy_b128(ksb + (unsigned)(r * KSP + seg * 8) * 2, gp);
    }
    #pragma unroll
    for (int g = 0; g < 2; ++g) {       // V tile (transposed): 64 rows x 64B
        int ch = tid + g * 128;
        int d = ch >> 2, seg = ch & 3;
        const unsigned short* gp =
            vbT + (((size_t)b * H_ + h) * HD_ + d) * SKV_ + kv0 + seg * 8;
        async_copy_b128(vsb + (unsigned)(d * AP + seg * 8) * 2, gp);
    }
}

__global__ __launch_bounds__(128)
void attn_wmma(const unsigned short* __restrict__ qb,
               const unsigned short* __restrict__ kb,
               const unsigned short* __restrict__ vbT,
               const float* __restrict__ mask,   // [B,SQ,SKV]
               float* __restrict__ ctx)          // [B,SQ,DIM]
{
    __shared__ __align__(16) unsigned short Ks[2][AT_KT][KSP];
    __shared__ __align__(16) unsigned short VsT[2][HD_][AP];
    __shared__ __align__(16) unsigned short Ps[4][16][AP];

    const int tid  = threadIdx.x;
    const int lane = tid & 31;
    const int wave = tid >> 5;
    const int h    = blockIdx.y;
    const int b    = blockIdx.z;
    const int q0   = blockIdx.x * AT_QT + wave * 16;
    const int hlf  = lane >> 4;
    const int l15  = lane & 15;
    const float scale = 0.125f;         // 1/sqrt(HD)

    // Q fragments: two 16x32 A-fragments over HD=64, contiguous 16B reads.
    Frag qf[2];
    {
        const unsigned short* qrow =
            qb + ((size_t)b * SQ_ + q0 + l15) * DIM_ + h * HD_;
        #pragma unroll
        for (int kc = 0; kc < 2; ++kc) {
            qf[kc].q[0] = *(const uint4*)(qrow + kc * 32 + 8 * hlf);
            qf[kc].q[1] = *(const uint4*)(qrow + kc * 32 + 16 + 8 * hlf);
        }
    }

    f32x8 accO[4] = {};
    float m_i[8], s_i[8];
    #pragma unroll
    for (int i = 0; i < 8; ++i) { m_i[i] = -1e30f; s_i[i] = 0.0f; }

    const float* mbase = mask + (size_t)b * SQ_ * (size_t)SKV_;
    const unsigned ksb0 = lds_off(&Ks[0][0][0]),  ksb1 = lds_off(&Ks[1][0][0]);
    const unsigned vsb0 = lds_off(&VsT[0][0][0]), vsb1 = lds_off(&VsT[1][0][0]);

    // Prologue: stage tile 0, then overlap next-tile async with compute.
    attn_stage(kb, vbT, b, h, 0, tid, ksb0, vsb0);
    wait_async0();
    __syncthreads();

    int p = 0;
    for (int kv0 = 0; kv0 < SKV_; kv0 += AT_KT) {
        if (kv0 + AT_KT < SKV_)
            attn_stage(kb, vbT, b, h, kv0 + AT_KT, tid,
                       p ? ksb0 : ksb1, p ? vsb0 : vsb1);

        // --- S = Q @ K^T : two 16x16 tiles, 2 k-steps each ---
        f32x8 s[2] = {};
        #pragma unroll
        for (int t = 0; t < 2; ++t) {
            #pragma unroll
            for (int kc = 0; kc < 2; ++kc) {
                Frag kf;
                const unsigned short* krow = &Ks[p][t * 16 + l15][0];
                kf.q[0] = *(const uint4*)(krow + kc * 32 + 8 * hlf);
                kf.q[1] = *(const uint4*)(krow + kc * 32 + 16 + 8 * hlf);
                s[t] = __builtin_amdgcn_wmma_f32_16x16x32_bf16(
                    false, qf[kc].v, false, kf.v, (short)0, s[t], false, false);
            }
        }

        // --- Scale + additive mask + online softmax ---
        float nm[8];
        #pragma unroll
        for (int i = 0; i < 8; ++i) {
            int qr = q0 + i + 8 * hlf;
            const float* mr = mbase + (size_t)qr * SKV_ + kv0;
            float v0 = s[0][i] * scale + mr[l15] * NEG_;
            float v1 = s[1][i] * scale + mr[16 + l15] * NEG_;
            s[0][i] = v0; s[1][i] = v1;
            float mx = fmaxf(v0, v1);
            mx = fmaxf(mx, __shfl_xor(mx, 1, 32));
            mx = fmaxf(mx, __shfl_xor(mx, 2, 32));
            mx = fmaxf(mx, __shfl_xor(mx, 4, 32));
            mx = fmaxf(mx, __shfl_xor(mx, 8, 32));
            nm[i] = fmaxf(m_i[i], mx);
        }
        #pragma unroll
        for (int i = 0; i < 8; ++i) {
            float p0 = __expf(s[0][i] - nm[i]);
            float p1 = __expf(s[1][i] - nm[i]);
            float alpha = __expf(m_i[i] - nm[i]);
            float ps = p0 + p1;
            ps += __shfl_xor(ps, 1, 32);
            ps += __shfl_xor(ps, 2, 32);
            ps += __shfl_xor(ps, 4, 32);
            ps += __shfl_xor(ps, 8, 32);
            s_i[i] = s_i[i] * alpha + ps;
            m_i[i] = nm[i];
            #pragma unroll
            for (int nt = 0; nt < 4; ++nt)
                accO[nt][i] *= alpha;
            Ps[wave][i + 8 * hlf][l15]      = f32_to_bf16(p0);
            Ps[wave][i + 8 * hlf][16 + l15] = f32_to_bf16(p1);
        }

        // --- P as 16x32 A-fragment (same-wave LDS RAW, HW-ordered) ---
        Frag pf;
        {
            const unsigned short* prow = &Ps[wave][l15][0];
            pf.q[0] = *(const uint4*)(prow + 8 * hlf);
            pf.q[1] = *(const uint4*)(prow + 16 + 8 * hlf);
        }

        // --- O += P @ V ---
        #pragma unroll
        for (int nt = 0; nt < 4; ++nt) {
            Frag vf;
            const unsigned short* vcol = &VsT[p][nt * 16 + l15][0];
            vf.q[0] = *(const uint4*)(vcol + 8 * hlf);
            vf.q[1] = *(const uint4*)(vcol + 16 + 8 * hlf);
            accO[nt] = __builtin_amdgcn_wmma_f32_16x16x32_bf16(
                false, pf.v, false, vf.v, (short)0, accO[nt], false, false);
        }

        wait_async0();          // next tile landed (this wave's copies)
        __syncthreads();        // all waves done copying + done reading tile p
        p ^= 1;
    }

    // Normalize, write ctx [B,SQ,DIM] fp32.
    #pragma unroll
    for (int i = 0; i < 8; ++i) {
        int qr = q0 + i + 8 * hlf;
        float inv = 1.0f / s_i[i];
        #pragma unroll
        for (int nt = 0; nt < 4; ++nt) {
            int d = nt * 16 + l15;
            ctx[((size_t)b * SQ_ + qr) * DIM_ + h * HD_ + d] = accO[nt][i] * inv;
        }
    }
}

// ---------------------------------------------------------------------------
// Cache copy: f32 cache -> f32 concat output + bf16 workspace copy.
// transposed=0: bf row-major [B,SKV,DIM]; =1: bf [B,H,HD,SKV] (for V).
// ---------------------------------------------------------------------------
__global__ __launch_bounds__(256)
void copy_cache(const float4* __restrict__ src, float4* __restrict__ dst,
                unsigned short* __restrict__ bf, int transposed)
{
    size_t i = (size_t)blockIdx.x * 256 + threadIdx.x;
    const size_t per_b_in  = (size_t)SC_ * DIM_ / 4;
    const size_t per_b_out = (size_t)SKV_ * DIM_ / 4;
    if (i >= (size_t)B_ * per_b_in) return;
    size_t bb = i / per_b_in, r = i % per_b_in;
    float4 f = src[i];
    dst[bb * per_b_out + r] = f;
    int kv = (int)(r / (DIM_ / 4));
    int c4 = (int)(r % (DIM_ / 4)) * 4;
    if (!transposed) {
        unsigned short* p = bf + (bb * SKV_ + kv) * (size_t)DIM_ + c4;
        cvt2_store(p,     f.x, f.y);
        cvt2_store(p + 2, f.z, f.w);
    } else {
        int hh = c4 >> 6, dd = c4 & 63;
        unsigned short e[4] = { f32_to_bf16(f.x), f32_to_bf16(f.y),
                                f32_to_bf16(f.z), f32_to_bf16(f.w) };
        #pragma unroll
        for (int j = 0; j < 4; ++j)
            bf[((bb * H_ + hh) * (size_t)HD_ + dd + j) * SKV_ + kv] = e[j];
    }
}

extern "C" void kernel_launch(void* const* d_in, const int* in_sizes, int n_in,
                              void* d_out, int out_size, void* d_ws, size_t ws_size,
                              hipStream_t stream) {
    const float* x       = (const float*)d_in[0];
    const float* k_cache = (const float*)d_in[1];
    const float* v_cache = (const float*)d_in[2];
    const float* mask    = (const float*)d_in[3];
    const float* Wq      = (const float*)d_in[4];
    const float* bq      = (const float*)d_in[5];
    const float* Wk      = (const float*)d_in[6];
    const float* bk      = (const float*)d_in[7];
    const float* Wv      = (const float*)d_in[8];
    const float* bv      = (const float*)d_in[9];
    const float* Wo      = (const float*)d_in[10];
    const float* bo      = (const float*)d_in[11];

    float* out  = (float*)d_out;                       // [B,SQ,DIM]
    float* kout = out  + (size_t)B_ * SQ_  * DIM_;     // [B,SKV,DIM]
    float* vout = kout + (size_t)B_ * SKV_ * DIM_;     // [B,SKV,DIM]

    // Workspace: bf16 Q/K/V(T) + fp32 ctx  (8 + 16 + 16 + 16 MB = 56 MB)
    unsigned short* qbw = (unsigned short*)d_ws;                  // [B,SQ,DIM]
    unsigned short* kbw = qbw + (size_t)B_ * SQ_  * DIM_;         // [B,SKV,DIM]
    unsigned short* vbT = kbw + (size_t)B_ * SKV_ * DIM_;         // [B,H,HD,SKV]
    float*          ctx = (float*)(vbT + (size_t)B_ * SKV_ * DIM_);

    const int M = B_ * SQ_, N = DIM_, K = DIM_;
    dim3 gg(N / GT_N, M / GT_M);

    // 1. Caches -> concat outputs + bf16 workspace copies.
    {
        size_t n4 = (size_t)B_ * SC_ * DIM_ / 4;
        dim3 gc((unsigned)((n4 + 255) / 256));
        copy_cache<<<gc, 256, 0, stream>>>((const float4*)k_cache, (float4*)kout, kbw, 0);
        copy_cache<<<gc, 256, 0, stream>>>((const float4*)v_cache, (float4*)vout, vbT, 1);
    }

    // 2. QKV projections. Q: bf16-only. K/V: f32 concat rows + bf16 copies.
    gemm_bf16_wmma<<<gg, 128, 0, stream>>>(x, Wq, bq, nullptr, qbw, M, N, K, SQ_,  0,   0);
    gemm_bf16_wmma<<<gg, 128, 0, stream>>>(x, Wk, bk, kout,    kbw, M, N, K, SKV_, SC_, 0);
    gemm_bf16_wmma<<<gg, 128, 0, stream>>>(x, Wv, bv, vout,    vbT, M, N, K, SKV_, SC_, 1);

    // 3. Attention (bf16 WMMA, double-buffered async-LDS staging).
    dim3 ga(SQ_ / AT_QT, H_, B_);
    attn_wmma<<<ga, 128, 0, stream>>>(qbw, kbw, vbT, mask, ctx);

    // 4. Output projection (fp32 result only).
    gemm_bf16_wmma<<<gg, 128, 0, stream>>>(ctx, Wo, bo, out, nullptr, M, N, K, SQ_, 0, 0);
}
